// DenseGridNet_3255585210920
// MI455X (gfx1250) — matrix operand order repaired and checked by reading, not compiled, not run on hardware
//
#include <hip/hip_runtime.h>

#define GRX 1024
#define GRY 1024
#define HID 64

typedef __attribute__((ext_vector_type(16))) _Float16 v16h;
typedef __attribute__((ext_vector_type(8)))  _Float16 v8h;
typedef __attribute__((ext_vector_type(8)))  float    v8f;

union HFrag { v16h v; _Float16 e[16]; };
union BFrag { v16h v; unsigned u[8]; };
union DFrag { v8h  v; _Float16 e[8]; unsigned u[4]; };

#define SWAPX16 0x401F  // ds_swizzle group-of-32: xor=0x10, or=0, and=0x1F

__device__ __forceinline__ unsigned swz16(unsigned v) {
    return (unsigned)__builtin_amdgcn_ds_swizzle((int)v, SWAPX16);
}
__device__ __forceinline__ float swz16f(float v) {
    return __builtin_bit_cast(float,
        (unsigned)__builtin_amdgcn_ds_swizzle(__builtin_bit_cast(int, v), SWAPX16));
}

__device__ __forceinline__ v8f wmma_f32f16(v16h a, v16h b, v8f c) {
    // D(f32) = A(16x32 f16) * B(32x16 f16) + C(16x16 f32)
    return __builtin_amdgcn_wmma_f32_16x16x32_f16(
        false, a, false, b, (short)0, c, false, false);
}
__device__ __forceinline__ v8h wmma_f16f16(v16h a, v16h b, v8h c) {
    // D(f16) = A(16x32 f16) * B(32x16 f16) + C(16x16 f16): 4-VGPR C/D
    return __builtin_amdgcn_wmma_f16_16x16x32_f16(
        false, a, false, b, (short)0, c, false, false);
}
// Packed relu: exactly one v_pk_max_num_f16 per dword (no canonicalize pair
// that llvm.maxnum inserts).
__device__ __forceinline__ v8h relu8h(v8h x) {
    DFrag d; d.v = x;
    #pragma unroll
    for (int j = 0; j < 4; ++j) {
        unsigned r;
        asm("v_pk_max_num_f16 %0, %1, 0" : "=v"(r) : "v"(d.u[j]));
        d.u[j] = r;
    }
    return d.v;
}

// A-fragment (16x32 f16) of W^T tile: element (M=o_local, K=i) =
// W[kbase+i][ocol0+o_local]; rows >= kvalid are 0 except optional bias row.
// Layout: lane (p,hi): VGPR j: K = 16*hi + 2*j+{0,1}.
__device__ __forceinline__ v16h load_w_frag(const float* __restrict__ W, int ldw,
                                            int kbase, int kvalid,
                                            int ocol0, int ocols,
                                            const float* __restrict__ bias, int biasrow,
                                            int p, int hi) {
    HFrag f;
    const int o = ocol0 + p;
    #pragma unroll
    for (int j = 0; j < 8; ++j) {
        const int i0 = kbase + 16 * hi + 2 * j;
        float a0 = 0.f, a1 = 0.f;
        if (p < ocols) {
            const int ia = i0, ib = i0 + 1;
            if (ia < kvalid)                a0 = W[ia * ldw + o];
            else if (bias && ia == biasrow) a0 = bias[o];
            if (ib < kvalid)                a1 = W[ib * ldw + o];
            else if (bias && ib == biasrow) a1 = bias[o];
        }
        f.e[2 * j]     = (_Float16)a0;
        f.e[2 * j + 1] = (_Float16)a1;
    }
    return f.v;
}

// f32 C-fragment bias: element (row = r + 8*hi, col = p) = b[base + row].
__device__ __forceinline__ v8f load_bias_frag_f32(const float* __restrict__ b,
                                                  int base, int nvalid, int hi) {
    v8f c;
    #pragma unroll
    for (int r = 0; r < 8; ++r) {
        const int idx = base + r + 8 * hi;
        c[r] = (idx < nvalid) ? b[idx] : 0.f;
    }
    return c;
}
// f16 C-fragment bias: lane (p,hi): element k (k=0..7) = b[base + k + 8*hi].
__device__ __forceinline__ v8h load_bias_frag_f16(const float* __restrict__ b,
                                                  int base, int hi) {
    DFrag c;
    #pragma unroll
    for (int k = 0; k < 8; ++k) c.e[k] = (_Float16)b[base + k + 8 * hi];
    return c.v;
}

// Convert a pair of f16 D tiles (channels [32s,32s+16) / [+16,+32), points in
// columns) into the next layer's B fragment (32x16 f16). The f16 C/D layout is
// already pair-packed, so this is a pure half-wave dword swap: 8 ds_swizzle,
// zero conversions.
//   B.u[j]   (rows 2j,2j+1 of tile 2s+hi, lane p)   = hi ? swz(u1[j]) : u0[j]
//   B.u[j+4] (rows 2j,2j+1 via lane p+16 half)      = hi ? u1[j]      : swz(u0[j])
__device__ __forceinline__ v16h dpair_to_bfrag(v8h d0, v8h d1, int hi) {
    DFrag a, b;
    a.v = d0; b.v = d1;
    BFrag f;
    #pragma unroll
    for (int j = 0; j < 4; ++j) {
        const unsigned u0x = swz16(a.u[j]);
        const unsigned u1x = swz16(b.u[j]);
        f.u[j]     = hi ? u1x    : a.u[j];
        f.u[j + 4] = hi ? b.u[j] : u0x;
    }
    return f.v;
}

__global__ __launch_bounds__(256)
void densegrid_mlp_fused(const float* __restrict__ x,   // (N,3)
                         const float* __restrict__ emb, // (RX*RY,4)
                         const float* __restrict__ w1, const float* __restrict__ b1,
                         const float* __restrict__ w2, const float* __restrict__ b2,
                         const float* __restrict__ w3, const float* __restrict__ b3,
                         float* __restrict__ out, int n) {
    const int lane = (int)(threadIdx.x & 31u);
    const int p    = lane & 15;
    const int hi   = lane >> 4;
    const int wav  = (int)(threadIdx.x >> 5);
    const int gw   = (int)blockIdx.x * ((int)blockDim.x >> 5) + wav;
    const int nw   = (int)gridDim.x * ((int)blockDim.x >> 5);

    // ---- Preload weights as WMMA A-fragments (register resident) ----
    v16h w1f[4];                                  // W1^T, bias folded at K=5
    #pragma unroll
    for (int t = 0; t < 4; ++t)
        w1f[t] = load_w_frag(w1, HID, 0, 5, 16 * t, 16, b1, 5, p, hi);
    v16h w2f[8];                                  // W2^T: 4 N-tiles x 2 K-steps
    #pragma unroll
    for (int t = 0; t < 4; ++t)
        #pragma unroll
        for (int s = 0; s < 2; ++s)
            w2f[t * 2 + s] = load_w_frag(w2, HID, 32 * s, 64, 16 * t, 16, nullptr, -1, p, hi);
    v16h w3f[2];                                  // W3^T: 1 N-tile x 2 K-steps
    #pragma unroll
    for (int s = 0; s < 2; ++s)
        w3f[s] = load_w_frag(w3, 3, 32 * s, 64, 0, 3, nullptr, -1, p, hi);
    v8h b2h[4];
    #pragma unroll
    for (int t = 0; t < 4; ++t) b2h[t] = load_bias_frag_f16(b2, 16 * t, hi);
    v8f b3f = load_bias_frag_f32(b3, 0, 3, hi);
    v8h zh;
    #pragma unroll
    for (int r = 0; r < 8; ++r) zh[r] = (_Float16)0.f;

    const float4* __restrict__ e4 = (const float4*)emb;

    for (int base = gw * 32; base < n; base += nw * 32) {
        const int nb = base + nw * 32;
        if (nb < n) __builtin_prefetch(x + (size_t)nb * 3 + lane, 0, 1);

        // ---- Per-lane: features of point base+lane (32 points / wave) ----
        int q = base + lane;
        if (q >= n) q = n - 1;
        const float3 xv = ((const float3*)x)[q];
        const float fu = xv.y * (float)GRX;
        int x0 = (int)fu; if (x0 == GRX) x0 = 0;
        const int x1 = (x0 + 1 == GRX) ? (GRX - 1) : (x0 + 1);
        const float wx = fu - (float)x0;
        const float fv = xv.z * (float)GRY;
        const int y0 = (int)fv;
        const int y1 = (y0 + 1 == GRY) ? (GRY - 1) : (y0 + 1);
        const float wy = fv - (float)y0;

        const float4 v00 = e4[y0 * GRX + x0];
        const float4 v10 = e4[y0 * GRX + x1];
        const float4 v01 = e4[y1 * GRX + x0];
        const float4 v11 = e4[y1 * GRX + x1];
        const float wxm = 1.f - wx, wym = 1.f - wy;

        float f[5];
        f[0] = xv.x;
        f[1] = (v00.x * wxm + v10.x * wx) * wym + (v01.x * wxm + v11.x * wx) * wy;
        f[2] = (v00.y * wxm + v10.y * wx) * wym + (v01.y * wxm + v11.y * wx) * wy;
        f[3] = (v00.z * wxm + v10.z * wx) * wym + (v01.z * wxm + v11.z * wx) * wy;
        f[4] = (v00.w * wxm + v10.w * wx) * wym + (v01.w * wxm + v11.w * wx) * wy;

        // ---- Two 16-point tiles per wave ----
        #pragma unroll
        for (int pt = 0; pt < 2; ++pt) {
            // Features of this tile's 16 points into lanes 0..15.
            float g[5];
            #pragma unroll
            for (int k = 0; k < 5; ++k)
                g[k] = pt ? swz16f(f[k]) : f[k];

            // Layer-1 B fragment: K<6 valid, constant-1 row drives the bias.
            HFrag bf;
            #pragma unroll
            for (int k = 0; k < 16; ++k) bf.e[k] = (_Float16)0.f;
            if (hi == 0) {
                #pragma unroll
                for (int k = 0; k < 5; ++k) bf.e[k] = (_Float16)g[k];
                bf.e[5] = (_Float16)1.f;
            }

            // Layer 1: H1(64x16, f16 D) = relu(W1^T @ F)
            v8h h1[4];
            #pragma unroll
            for (int t = 0; t < 4; ++t)
                h1[t] = relu8h(wmma_f16f16(w1f[t], bf.v, zh));

            // Layer 2: H2(f16 D) = relu(W2^T @ H1 + b2)
            const v16h bs0 = dpair_to_bfrag(h1[0], h1[1], hi);
            const v16h bs1 = dpair_to_bfrag(h1[2], h1[3], hi);
            v8h h2[4];
            #pragma unroll
            for (int t = 0; t < 4; ++t) {
                v8h c = wmma_f16f16(w2f[t * 2 + 0], bs0, b2h[t]);
                h2[t] = relu8h(wmma_f16f16(w2f[t * 2 + 1], bs1, c));
            }

            // Layer 3 (f32 accumulate): O = sigmoid(W3^T @ H2 + b3)
            const v16h cs0 = dpair_to_bfrag(h2[0], h2[1], hi);
            const v16h cs1 = dpair_to_bfrag(h2[2], h2[3], hi);
            v8f o = wmma_f32f16(w3f[0], cs0, b3f);
            o = wmma_f32f16(w3f[1], cs1, o);

            // Rows 0..2 (lanes 0..15) hold rgb of point base + 16*pt + p.
            if (hi == 0) {
                const int qq = base + pt * 16 + p;
                if (qq < n) {
                    float3 rgb;
                    rgb.x = 1.f / (1.f + __expf(-o[0]));
                    rgb.y = 1.f / (1.f + __expf(-o[1]));
                    rgb.z = 1.f / (1.f + __expf(-o[2]));
                    ((float3*)out)[qq] = rgb;
                }
            }
        }
    }
}

extern "C" void kernel_launch(void* const* d_in, const int* in_sizes, int n_in,
                              void* d_out, int out_size, void* d_ws, size_t ws_size,
                              hipStream_t stream) {
    const float* x   = (const float*)d_in[0];
    const float* emb = (const float*)d_in[1];
    const float* w1  = (const float*)d_in[2];
    const float* b1  = (const float*)d_in[3];
    const float* w2  = (const float*)d_in[4];
    const float* b2  = (const float*)d_in[5];
    const float* w3  = (const float*)d_in[6];
    const float* b3  = (const float*)d_in[7];
    float* out = (float*)d_out;
    const int n = in_sizes[0] / 3;

    dim3 grid(2048), block(256); // 8 wave32 per block; grid-stride, 32 pts/wave/iter
    hipLaunchKernelGGL(densegrid_mlp_fused, grid, block, 0, stream,
                       x, emb, w1, b1, w2, b2, w3, b3, out, n);
}
